// Model_68762426409614
// MI455X (gfx1250) — matrix operand order repaired and checked by reading, not compiled, not run on hardware
//
#include <hip/hip_runtime.h>

typedef __bf16 v16bf __attribute__((ext_vector_type(16)));
typedef float  v8f   __attribute__((ext_vector_type(8)));

static const int NU_  = 100000;
static const int NM_  = 50000;
static const int EMM_ = 2000000;
static const int EMU_ = 2000000;
static const int L_   = 1000000;

// ---------------------------------------------------------------------------
// bf16 operand packing for V_WMMA_F32_16X16X32_BF16 (wave32)
//
// A (16x32, row major in memory, ld=64):
//   lanes 0-15  : row M=lane,    elems 0..15 = K[kc+0..15]
//   lanes 16-31 : row M=lane-16, elems 0..15 = K[kc+16..31]
// B (32x16) built from weight W stored [out][in] row-major (computing X@W^T,
// so B[k][n] = W[n][k]):
//   lanes 0-15  : col N=lane,    elems 0..15 = K[kc+0..15]
//   lanes 16-31 : col N=lane-16, elems 0..15 = K[kc+16..31]
// ---------------------------------------------------------------------------
__device__ __forceinline__ v16bf pack_a64(const float* __restrict__ base,
                                          int row, int kc, int lane) {
  const float* p0 = base + (size_t)row * 64 + kc + ((lane & 16) ? 16 : 0);
  const float* p1 = p0 + 8;
  v16bf a;
#pragma unroll
  for (int i = 0; i < 8; ++i) a[i] = (__bf16)p0[i];
#pragma unroll
  for (int i = 0; i < 8; ++i) a[8 + i] = (__bf16)p1[i];
  return a;
}

__device__ __forceinline__ v16bf pack_bT(const float* __restrict__ W, int ld,
                                         int nb, int kc, int lane) {
  const float* p = W + (size_t)(nb + (lane & 15)) * ld + kc + ((lane & 16) ? 16 : 0);
  v16bf b;
#pragma unroll
  for (int i = 0; i < 16; ++i) b[i] = (__bf16)p[i];
  return b;
}

// ---------------------------------------------------------------------------
// Y[N,64] = act( A1 @ W1^T + bias + (HAS_A2 ? gather(A2,idx2) @ W2^T : 0) )
// One wave per 16-row tile (grid-stride); weights register-resident.
// ---------------------------------------------------------------------------
template <bool HAS_A2, bool RELU>
__global__ __launch_bounds__(256) void wmma_dual_linear(
    const float* __restrict__ A1,
    const float* __restrict__ A2, const int* __restrict__ idx2,
    const float* __restrict__ W1, const float* __restrict__ W2,
    const float* __restrict__ bias, float* __restrict__ Y, int N)
{
  const int lane   = threadIdx.x & 31;
  const int wave   = (blockIdx.x * blockDim.x + threadIdx.x) >> 5;
  const int nwaves = (gridDim.x * blockDim.x) >> 5;
  const int tiles  = N >> 4;
  const int ncol   = lane & 15;
  const int mofs   = (lane & 16) ? 8 : 0;

  v16bf b1[2][4], b2r[2][4];
#pragma unroll
  for (int kc = 0; kc < 2; ++kc)
#pragma unroll
    for (int g = 0; g < 4; ++g) {
      b1[kc][g] = pack_bT(W1, 64, g * 16, kc * 32, lane);
      if (HAS_A2) b2r[kc][g] = pack_bT(W2, 64, g * 16, kc * 32, lane);
    }

  for (int t = wave; t < tiles; t += nwaves) {
    const int rowbase = t << 4;
    const int rlane   = rowbase + ncol;
    v8f c[4] = {};
#pragma unroll
    for (int kc = 0; kc < 2; ++kc) {
      v16bf a = pack_a64(A1, rlane, kc * 32, lane);
#pragma unroll
      for (int g = 0; g < 4; ++g)
        c[g] = __builtin_amdgcn_wmma_f32_16x16x32_bf16(
            false, a, false, b1[kc][g], (short)0, c[g], false, false);
    }
    if (HAS_A2) {
      const int r2 = idx2 ? idx2[rlane] : rlane;
#pragma unroll
      for (int kc = 0; kc < 2; ++kc) {
        v16bf a = pack_a64(A2, r2, kc * 32, lane);
#pragma unroll
        for (int g = 0; g < 4; ++g)
          c[g] = __builtin_amdgcn_wmma_f32_16x16x32_bf16(
              false, a, false, b2r[kc][g], (short)0, c[g], false, false);
      }
    }
#pragma unroll
    for (int g = 0; g < 4; ++g) {
      const float bv = bias[g * 16 + ncol];
#pragma unroll
      for (int r = 0; r < 8; ++r) {
        float v = c[g][r] + bv;
        if (RELU) v = fmaxf(v, 0.0f);
        // lanes 0-15 / 16-31 each write a contiguous 64B row segment
        Y[(size_t)(rowbase + mofs + r) * 64 + g * 16 + ncol] = v;
      }
    }
  }
}

// ---------------------------------------------------------------------------
// Edge decoder, fully fused:
// out[e] = relu(concat(zu[row[e]], zm[col[e]]) @ W1^T + b1) . W2 + b2
// ---------------------------------------------------------------------------
__global__ __launch_bounds__(256) void wmma_edge_decoder(
    const float* __restrict__ zu, const float* __restrict__ zm,
    const int* __restrict__ erow, const int* __restrict__ ecol,
    const float* __restrict__ W1 /*64x128*/, const float* __restrict__ b1,
    const float* __restrict__ W2 /*64*/, const float* __restrict__ b2,
    float* __restrict__ out, int L)
{
  const int lane   = threadIdx.x & 31;
  const int wave   = (blockIdx.x * blockDim.x + threadIdx.x) >> 5;
  const int nwaves = (gridDim.x * blockDim.x) >> 5;
  const int tiles  = L >> 4;
  const int ncol   = lane & 15;
  const int mofs   = (lane & 16) ? 8 : 0;

  v16bf bu[2][4], bm[2][4];
#pragma unroll
  for (int kc = 0; kc < 2; ++kc)
#pragma unroll
    for (int g = 0; g < 4; ++g) {
      bu[kc][g] = pack_bT(W1,      128, g * 16, kc * 32, lane);  // z_user half
      bm[kc][g] = pack_bT(W1 + 64, 128, g * 16, kc * 32, lane);  // z_movie half
    }
  float b1v[4], w2v[4];
#pragma unroll
  for (int g = 0; g < 4; ++g) {
    b1v[g] = b1[g * 16 + ncol];
    w2v[g] = W2[g * 16 + ncol];
  }
  const float b2v = b2[0];

  for (int t = wave; t < tiles; t += nwaves) {
    const int rowbase = t << 4;
    const int e = rowbase + ncol;
    const int u = erow[e];
    const int m = ecol[e];
    v8f c[4] = {};
#pragma unroll
    for (int kc = 0; kc < 2; ++kc) {
      v16bf a = pack_a64(zu, u, kc * 32, lane);
#pragma unroll
      for (int g = 0; g < 4; ++g)
        c[g] = __builtin_amdgcn_wmma_f32_16x16x32_bf16(
            false, a, false, bu[kc][g], (short)0, c[g], false, false);
    }
#pragma unroll
    for (int kc = 0; kc < 2; ++kc) {
      v16bf a = pack_a64(zm, m, kc * 32, lane);
#pragma unroll
      for (int g = 0; g < 4; ++g)
        c[g] = __builtin_amdgcn_wmma_f32_16x16x32_bf16(
            false, a, false, bm[kc][g], (short)0, c[g], false, false);
    }
    // 64 -> 1 projection in registers + cross-lane reduce over 16-lane halves
#pragma unroll
    for (int r = 0; r < 8; ++r) {
      float p = 0.0f;
#pragma unroll
      for (int g = 0; g < 4; ++g) {
        float h = fmaxf(c[g][r] + b1v[g], 0.0f);
        p = fmaf(h, w2v[g], p);
      }
      p += __shfl_xor(p, 1, 32);
      p += __shfl_xor(p, 2, 32);
      p += __shfl_xor(p, 4, 32);
      p += __shfl_xor(p, 8, 32);
      if (ncol == 0) out[rowbase + mofs + r] = p + b2v;
    }
  }
}

// ---------------------------------------------------------------------------
// Segment-mean helpers. In seg_scatter each wave32 covers 32 consecutive
// features of ONE edge, so the edge indices are wave-uniform: force scalar
// loads with readfirstlane; the 32 atomics hit one contiguous 128B chunk.
// ---------------------------------------------------------------------------
__global__ __launch_bounds__(256) void seg_count_kernel(
    const int* __restrict__ dst, int E, float* __restrict__ cnt)
{
  for (int i = blockIdx.x * blockDim.x + threadIdx.x; i < E;
       i += gridDim.x * blockDim.x)
    atomicAdd(&cnt[dst[i]], 1.0f);
}

__global__ __launch_bounds__(256) void seg_scatter_kernel(
    const float* __restrict__ x, const int* __restrict__ src,
    const int* __restrict__ dst, int E, float* __restrict__ agg)
{
  const int total = E << 6;  // E * 64 (< 2^31)
  for (int i = blockIdx.x * blockDim.x + threadIdx.x; i < total;
       i += gridDim.x * blockDim.x) {
    const int e = __builtin_amdgcn_readfirstlane(i >> 6);  // wave-uniform
    const int f = i & 63;
    const int d = __builtin_amdgcn_readfirstlane(dst[e]);
    const int s = __builtin_amdgcn_readfirstlane(src[e]);
    atomicAdd(&agg[((size_t)d << 6) + f], x[((size_t)s << 6) + f]);
  }
}

__global__ __launch_bounds__(256) void seg_div_kernel(
    float* __restrict__ agg, const float* __restrict__ cnt, int N)
{
  const int total = N << 6;
  for (int i = blockIdx.x * blockDim.x + threadIdx.x; i < total;
       i += gridDim.x * blockDim.x)
    agg[i] /= fmaxf(cnt[i >> 6], 1.0f);
}

static inline int imin_(int a, int b) { return a < b ? a : b; }

extern "C" void kernel_launch(void* const* d_in, const int* in_sizes, int n_in,
                              void* d_out, int out_size, void* d_ws, size_t ws_size,
                              hipStream_t stream) {
  (void)in_sizes; (void)n_in; (void)out_size; (void)ws_size;
  const float* x_movie  = (const float*)d_in[0];
  const int*   user_ids = (const int*)d_in[1];
  const int*   e_mm     = (const int*)d_in[2];   // [2, EMM]
  const int*   e_mu     = (const int*)d_in[3];   // [2, EMU]
  const int*   eli      = (const int*)d_in[4];   // [2, L]
  const float* user_emb = (const float*)d_in[5];
  const float* ue_c1_Wl = (const float*)d_in[6];
  const float* ue_c1_bl = (const float*)d_in[7];
  const float* ue_c2_Wl = (const float*)d_in[8];
  const float* ue_c2_bl = (const float*)d_in[9];
  const float* ue_c3_Wl = (const float*)d_in[10];
  const float* ue_c3_bl = (const float*)d_in[11];
  const float* ue_c1_Wr = (const float*)d_in[12];
  const float* ue_c2_Wr = (const float*)d_in[13];
  const float* ue_c3_Wr = (const float*)d_in[14];
  const float* ue_lin_W = (const float*)d_in[15];
  const float* ue_lin_b = (const float*)d_in[16];
  const float* me_c1_Wl = (const float*)d_in[17];
  const float* me_c1_bl = (const float*)d_in[18];
  const float* me_c1_Wr = (const float*)d_in[19];
  const float* me_c2_Wl = (const float*)d_in[20];
  const float* me_c2_bl = (const float*)d_in[21];
  const float* me_c2_Wr = (const float*)d_in[22];
  const float* me_lin_W = (const float*)d_in[23];
  const float* me_lin_b = (const float*)d_in[24];
  const float* dec_W1   = (const float*)d_in[25];
  const float* dec_b1   = (const float*)d_in[26];
  const float* dec_W2   = (const float*)d_in[27];
  const float* dec_b2   = (const float*)d_in[28];

  // ---- workspace carve -----------------------------------------------------
  char* w = (char*)d_ws;
  auto alloc = [&](size_t bytes) {
    char* p = w;
    w += (bytes + 255) & ~(size_t)255;
    return p;
  };
  float* cnt_mm = (float*)alloc((size_t)NM_ * 4);
  float* cnt_mu = (float*)alloc((size_t)NU_ * 4);
  float* bufA = (float*)alloc((size_t)NM_ * 64 * 4);  // movie-side aggregates
  float* bufC = (float*)alloc((size_t)NM_ * 64 * 4);  // movie_x, later m2
  float* bufD = (float*)alloc((size_t)NM_ * 64 * 4);  // m1, later z_movie
  float* bufB = (float*)alloc((size_t)NU_ * 64 * 4);  // user-side aggregates
  float* bufE = (float*)alloc((size_t)NU_ * 64 * 4);  // user_x1, later z_user
  float* bufF = (float*)alloc((size_t)NU_ * 64 * 4);  // user_x2

  const int gNM = imin_(((NM_ / 16) + 7) / 8, 8192);   // 8 waves per block
  const int gNU = imin_(((NU_ / 16) + 7) / 8, 8192);
  const int gL  = imin_(((L_  / 16) + 7) / 8, 16384);

  // ---- stage 1: first-hop aggregations ------------------------------------
  hipMemsetAsync(cnt_mm, 0, (size_t)NM_ * 4, stream);
  hipMemsetAsync(cnt_mu, 0, (size_t)NU_ * 4, stream);
  hipMemsetAsync(bufA, 0, (size_t)NM_ * 64 * 4, stream);
  hipMemsetAsync(bufB, 0, (size_t)NU_ * 64 * 4, stream);
  seg_count_kernel<<<2048, 256, 0, stream>>>(e_mm + EMM_, EMM_, cnt_mm);
  seg_count_kernel<<<2048, 256, 0, stream>>>(e_mu + EMU_, EMU_, cnt_mu);
  seg_scatter_kernel<<<8192, 256, 0, stream>>>(x_movie, e_mm, e_mm + EMM_, EMM_, bufA);
  seg_scatter_kernel<<<8192, 256, 0, stream>>>(x_movie, e_mu, e_mu + EMU_, EMU_, bufB);
  seg_div_kernel<<<4096, 256, 0, stream>>>(bufA, cnt_mm, NM_);
  seg_div_kernel<<<4096, 256, 0, stream>>>(bufB, cnt_mu, NU_);

  // ---- stage 2: first conv layers (WMMA) ----------------------------------
  wmma_dual_linear<true, true><<<gNM, 256, 0, stream>>>(bufA, x_movie, nullptr,
      ue_c1_Wl, ue_c1_Wr, ue_c1_bl, bufC, NM_);                     // movie_x
  wmma_dual_linear<true, true><<<gNM, 256, 0, stream>>>(bufA, x_movie, nullptr,
      me_c1_Wl, me_c1_Wr, me_c1_bl, bufD, NM_);                     // m1
  wmma_dual_linear<true, true><<<gNU, 256, 0, stream>>>(bufB, user_emb, user_ids,
      ue_c2_Wl, ue_c2_Wr, ue_c2_bl, bufE, NU_);                     // user_x1

  // ---- stage 3: second-hop aggregations -----------------------------------
  hipMemsetAsync(bufA, 0, (size_t)NM_ * 64 * 4, stream);
  hipMemsetAsync(bufB, 0, (size_t)NU_ * 64 * 4, stream);
  seg_scatter_kernel<<<8192, 256, 0, stream>>>(bufC /*movie_x*/, e_mu, e_mu + EMU_, EMU_, bufB);
  seg_scatter_kernel<<<8192, 256, 0, stream>>>(bufD /*m1*/,      e_mm, e_mm + EMM_, EMM_, bufA);
  seg_div_kernel<<<4096, 256, 0, stream>>>(bufA, cnt_mm, NM_);
  seg_div_kernel<<<4096, 256, 0, stream>>>(bufB, cnt_mu, NU_);

  // ---- stage 4: second conv + output projections --------------------------
  wmma_dual_linear<true, true><<<gNU, 256, 0, stream>>>(bufB, bufE, nullptr,
      ue_c3_Wl, ue_c3_Wr, ue_c3_bl, bufF, NU_);                     // user_x2
  wmma_dual_linear<false, false><<<gNU, 256, 0, stream>>>(bufF, nullptr, nullptr,
      ue_lin_W, nullptr, ue_lin_b, bufE, NU_);                      // z_user
  wmma_dual_linear<true, true><<<gNM, 256, 0, stream>>>(bufA, bufD, nullptr,
      me_c2_Wl, me_c2_Wr, me_c2_bl, bufC, NM_);                     // m2
  wmma_dual_linear<false, false><<<gNM, 256, 0, stream>>>(bufC, nullptr, nullptr,
      me_lin_W, nullptr, me_lin_b, bufD, NM_);                      // z_movie

  // ---- stage 5: fused edge decoder ----------------------------------------
  wmma_edge_decoder<<<gL, 256, 0, stream>>>(bufE, bufD, eli, eli + L_,
      dec_W1, dec_b1, dec_W2, dec_b2, (float*)d_out, L_);
}
